// GATEncoder_25872882991697
// MI455X (gfx1250) — matrix operand order repaired and checked by reading, not compiled, not run on hardware
//
#include <hip/hip_runtime.h>

typedef __bf16 bf16;
typedef bf16  bf16x8  __attribute__((ext_vector_type(8)));
typedef bf16  bf16x16 __attribute__((ext_vector_type(16)));
typedef float f32x8   __attribute__((ext_vector_type(8)));

constexpr int N_NODES = 10000;
constexpr int N_EDGES = 160000;
constexpr int ETOT    = N_EDGES + N_NODES;   // with self-loops
constexpr int F_IN    = 2000;
constexpr int KP1     = 2016;                // F_IN padded to multiple of 32
constexpr int HEADS   = 10;
constexpr int C       = 128;
constexpr int HC      = HEADS * C;           // 1280 (= GEMM Nc, compile-time)
constexpr int K2      = HC;                  // layer-2 K (already /32)

__device__ __forceinline__ unsigned short f2bf(float f) {
  unsigned u = __float_as_uint(f);
  u += 0x7fffu + ((u >> 16) & 1u);           // round-to-nearest-even
  return (unsigned short)(u >> 16);
}

// CDNA5 async global->LDS copy (16B per lane), tracked by ASYNCcnt.
// LDS generic addresses map to the LDS offset in addr[31:0] (ISA 10.2).
__device__ __forceinline__ void async_copy16(const unsigned short* gptr,
                                             unsigned short* lptr) {
  unsigned lofs = (unsigned)(unsigned long long)lptr;
  asm volatile("global_load_async_to_lds_b128 %0, %1, off"
               :: "v"(lofs), "v"(gptr) : "memory");
}
__device__ __forceinline__ void async_wait0() {
  asm volatile("s_wait_asynccnt 0x0" ::: "memory");
}
__device__ __forceinline__ void async_wait4() {   // allow one 4-op batch in flight
  asm volatile("s_wait_asynccnt 0x4" ::: "memory");
}

// ---------------------------------------------------------------- conversions
__global__ void convert_pad_bf16(const float* __restrict__ x,
                                 unsigned short* __restrict__ xb,
                                 int rows, int kin, int kpad) {
  int i = blockIdx.x * 256 + threadIdx.x;
  if (i >= rows * kpad) return;
  int k = i % kpad, n = i / kpad;
  float v = (k < kin) ? x[(size_t)n * kin + k] : 0.f;
  xb[i] = f2bf(v);
}

// WT[n][k] = W[k][n], bf16, K zero-padded to kpad
__global__ void transpose_w_bf16(const float* __restrict__ W,
                                 unsigned short* __restrict__ WT,
                                 int kin, int nout, int kpad) {
  int i = blockIdx.x * 256 + threadIdx.x;
  if (i >= nout * kpad) return;
  int k = i % kpad, n = i / kpad;
  float v = (k < kin) ? W[(size_t)k * nout + n] : 0.f;
  WT[i] = f2bf(v);
}

__global__ void bias_relu_bf16(const float* __restrict__ in,
                               const float* __restrict__ bias,
                               unsigned short* __restrict__ outb, int total) {
  int i = blockIdx.x * 256 + threadIdx.x;
  if (i >= total) return;
  float v = in[i] + bias[i % HC];
  outb[i] = f2bf(v > 0.f ? v : 0.f);
}

// ---------------------------------------------------------------- CSR build
__global__ void zero_i32(int* __restrict__ p, int n) {
  int i = blockIdx.x * 256 + threadIdx.x;
  if (i < n) p[i] = 0;
}

__global__ void csr_count(const int* __restrict__ ei, int* __restrict__ counts) {
  int e = blockIdx.x * 256 + threadIdx.x;
  if (e >= ETOT) return;
  int d = (e < N_EDGES) ? ei[N_EDGES + e] : (e - N_EDGES);
  atomicAdd(&counts[d], 1);
}

__global__ __launch_bounds__(1024)
void scan_fill(const int* __restrict__ counts, int* __restrict__ rowptr,
               int* __restrict__ fillpos, int n) {
  __shared__ int part[1024];
  constexpr int CH = (N_NODES + 1023) / 1024;
  int t = threadIdx.x;
  int base = t * CH;
  int s = 0;
  for (int j = 0; j < CH; ++j) { int i = base + j; if (i < n) s += counts[i]; }
  part[t] = s;
  __syncthreads();
  for (int off = 1; off < 1024; off <<= 1) {
    int v = part[t];
    int add = (t >= off) ? part[t - off] : 0;
    __syncthreads();
    part[t] = v + add;
    __syncthreads();
  }
  int run = part[t] - s;                      // exclusive prefix
  for (int j = 0; j < CH; ++j) {
    int i = base + j;
    if (i < n) { rowptr[i] = run; fillpos[i] = run; run += counts[i]; }
  }
  if (t == 1023) rowptr[n] = part[1023];
}

__global__ void csr_fill(const int* __restrict__ ei, int* __restrict__ fillpos,
                         int* __restrict__ csr_src) {
  int e = blockIdx.x * 256 + threadIdx.x;
  if (e >= ETOT) return;
  int s, d;
  if (e < N_EDGES) { s = ei[e]; d = ei[N_EDGES + e]; }
  else             { s = e - N_EDGES; d = s; }
  int pos = atomicAdd(&fillpos[d], 1);
  csr_src[pos] = s;
}

// ---------------------------------------------------------------- bf16 WMMA GEMM
// C[M,HC](f32) = A[M,ld](bf16 row-major) x B, with B given transposed:
// BT[HC,ld](bf16), BT[n][k] = B[k][n].
// Block tile 128x128, 8 waves; each wave owns 32x64 = 2x4 WMMA 16x16x32 tiles.
// Triple-buffered LDS filled by async global->LDS copies with pipeline depth 2:
// at step k we issue the batch for k+2 and only require batch k+1 complete
// (s_wait_asynccnt <= 4; async loads complete in order), giving a full k-step
// of L2 latency slack. No VGPR staging, no ds_store, no spills.
// EXEC stays full everywhere; row guards are uniform per 16-row subtile.
__global__ __launch_bounds__(256)
void wmma_gemm(const unsigned short* __restrict__ A,
               const unsigned short* __restrict__ BT,
               float* __restrict__ Cmat,
               int M, int ld, int ksteps) {
  __shared__ unsigned short As[3][128 * 40];   // 128 rows x 32 bf16, stride 40
  __shared__ unsigned short Bs[3][128 * 40];

  const int tid  = threadIdx.x;
  const int lane = tid & 31;
  const int wave = tid >> 5;
  const int wm   = wave & 3;                   // m quadrant -> *32
  const int wn   = wave >> 2;                  // n half     -> *64
  const int lrow = lane & 15;
  const int half = lane >> 4;

  const int m0 = blockIdx.x * 128;
  const int n0 = blockIdx.y * 128;

  f32x8 acc[2][4] = {};

  const int srow = tid >> 2;                   // staging rows srow, srow+64
  const int sgrp = tid & 3;                    // 8-bf16 group within row

  // async-stage one 128x32 A panel + 128x32 BT panel into LDS[buf] (4 ops/lane)
  auto aload = [&](int k0, int buf) {
    #pragma unroll
    for (int r = 0; r < 2; ++r) {
      int row = srow + r * 64;
      int gm  = m0 + row;
      int gmc = (gm < M) ? gm : 0;             // clamp OOB rows (stores are guarded)
      async_copy16(A  + (size_t)gmc * ld + k0 + sgrp * 8,
                   &As[buf][row * 40 + sgrp * 8]);
      async_copy16(BT + (size_t)(n0 + row) * ld + k0 + sgrp * 8,
                   &Bs[buf][row * 40 + sgrp * 8]);
    }
  };

  // prologue: batches 0 and 1 in flight; batch 0 must be complete
  aload(0, 0);
  if (ksteps > 1) { aload(32, 1); async_wait4(); }
  else            { async_wait0(); }
  __syncthreads();

  int cur = 0;
  for (int kk = 0; kk < ksteps; ++kk) {
    if (kk + 2 < ksteps) {
      int wbuf = cur + 2; if (wbuf >= 3) wbuf -= 3;
      aload((kk + 2) * 32, wbuf);              // overlap with WMMAs below
    }

    bf16x16 afrag[2], bfrag[4];
    #pragma unroll
    for (int mi = 0; mi < 2; ++mi) {
      // A 16x32 frag: lane (m=lrow, half): elems 0..7 = K half*8+0..7,
      // elems 8..15 = K half*8+16..23 -> two 16B LDS loads
      const unsigned short* p = &As[cur][(wm * 32 + mi * 16 + lrow) * 40 + half * 8];
      bf16x8 lo = *(const bf16x8*)(const void*)(p);
      bf16x8 hi = *(const bf16x8*)(const void*)(p + 16);
      afrag[mi] = __builtin_shufflevector(lo, hi, 0,1,2,3,4,5,6,7,8,9,10,11,12,13,14,15);
    }
    #pragma unroll
    for (int ni = 0; ni < 4; ++ni) {
      // B 32x16 frag: lane (n=lrow, half): elems 0..7 = K half*8+0..7,
      // elems 8..15 = K half*8+16..23 (same as A)
      const unsigned short* p = &Bs[cur][(wn * 64 + ni * 16 + lrow) * 40 + half * 8];
      bf16x8 lo = *(const bf16x8*)(const void*)(p);
      bf16x8 hi = *(const bf16x8*)(const void*)(p + 16);
      bfrag[ni] = __builtin_shufflevector(lo, hi, 0,1,2,3,4,5,6,7,8,9,10,11,12,13,14,15);
    }
    #pragma unroll
    for (int mi = 0; mi < 2; ++mi)
      #pragma unroll
      for (int ni = 0; ni < 4; ++ni)
        acc[mi][ni] = __builtin_amdgcn_wmma_f32_16x16x32_bf16(
            false, afrag[mi], false, bfrag[ni], (short)0, acc[mi][ni], false, false);

    // batch kk+1 must be complete before anyone reads its buffer
    if (kk + 2 < ksteps) async_wait4();
    else                 async_wait0();
    __syncthreads();      // everyone's writes visible; everyone's reads of cur done
    if (++cur >= 3) cur = 0;
  }

  // C/D layout: lane: n = lrow; VGPR r: m = r + 8*half.
  // Nc is compile-time (HC) so r*HC folds into immediate store offsets.
  #pragma unroll
  for (int mi = 0; mi < 2; ++mi) {
    const int sub0 = m0 + wm * 32 + mi * 16;
    #pragma unroll
    for (int ni = 0; ni < 4; ++ni) {
      const int col = n0 + wn * 64 + ni * 16 + lrow;
      float* p = Cmat + (size_t)(sub0 + half * 8) * HC + col;
      if (sub0 + 16 <= M) {                    // uniform: full subtile in range
        #pragma unroll
        for (int r = 0; r < 8; ++r) p[r * HC] = acc[mi][ni][r];
      } else if (sub0 < M) {                   // generic fallback (unused: M%16==0)
        #pragma unroll
        for (int r = 0; r < 8; ++r)
          if (sub0 + half * 8 + r < M) p[r * HC] = acc[mi][ni][r];
      }
    }
  }
}

// ---------------------------------------------------------------- attention
// one wave per (node, head): a_src/a_dst dot products over C=128
__global__ __launch_bounds__(256)
void att_dots(const float* __restrict__ h, const float* __restrict__ att_s,
              const float* __restrict__ att_d, float* __restrict__ a_s,
              float* __restrict__ a_d) {
  int gw   = (blockIdx.x * 256 + threadIdx.x) >> 5;
  int lane = threadIdx.x & 31;
  if (gw >= N_NODES * HEADS) return;
  int node = gw / HEADS, head = gw % HEADS;
  const float* hp = h + (size_t)node * HC + head * C;
  const float* as = att_s + head * C;
  const float* ad = att_d + head * C;
  float ps = 0.f, pd = 0.f;
  #pragma unroll
  for (int i = 0; i < 4; ++i) {
    int c = lane + i * 32;
    float v = hp[c];
    ps += v * as[c];
    pd += v * ad[c];
  }
  #pragma unroll
  for (int o = 16; o > 0; o >>= 1) {
    ps += __shfl_xor(ps, o, 32);
    pd += __shfl_xor(pd, o, 32);
  }
  if (lane == 0) { a_s[gw] = ps; a_d[gw] = pd; }
}

// one wave per (dst, head): segment softmax + weighted gather, no atomics
__global__ __launch_bounds__(256)
void gat_aggregate(const float* __restrict__ h, const float* __restrict__ a_s,
                   const float* __restrict__ a_d, const int* __restrict__ rowptr,
                   const int* __restrict__ csr_src, float* __restrict__ out) {
  int gw   = (blockIdx.x * 256 + threadIdx.x) >> 5;
  int lane = threadIdx.x & 31;
  if (gw >= N_NODES * HEADS) return;
  int node = gw / HEADS, head = gw % HEADS;
  int beg = rowptr[node], end = rowptr[node + 1];   // deg >= 1 (self-loop)
  float adst = a_d[node * HEADS + head];

  float mx = -3.0e38f;
  for (int i = beg + lane; i < end; i += 32) {
    float ev = a_s[csr_src[i] * HEADS + head] + adst;
    ev = (ev > 0.f) ? ev : 0.2f * ev;
    mx = fmaxf(mx, ev);
  }
  #pragma unroll
  for (int o = 16; o > 0; o >>= 1) mx = fmaxf(mx, __shfl_xor(mx, o, 32));

  float sm = 0.f;
  for (int i = beg + lane; i < end; i += 32) {
    float ev = a_s[csr_src[i] * HEADS + head] + adst;
    ev = (ev > 0.f) ? ev : 0.2f * ev;
    sm += __expf(ev - mx);
  }
  #pragma unroll
  for (int o = 16; o > 0; o >>= 1) sm += __shfl_xor(sm, o, 32);
  float inv = 1.f / (sm + 1e-16f);

  float a0 = 0.f, a1 = 0.f, a2 = 0.f, a3 = 0.f;
  for (int i = beg; i < end; ++i) {
    int s = csr_src[i];
    float ev = a_s[s * HEADS + head] + adst;
    ev = (ev > 0.f) ? ev : 0.2f * ev;
    float alpha = __expf(ev - mx) * inv;
    const float* hp = h + (size_t)s * HC + head * C;
    a0 += alpha * hp[lane];
    a1 += alpha * hp[lane + 32];
    a2 += alpha * hp[lane + 64];
    a3 += alpha * hp[lane + 96];
  }
  float* op = out + (size_t)node * HC + head * C;
  op[lane]      = a0;
  op[lane + 32] = a1;
  op[lane + 64] = a2;
  op[lane + 96] = a3;
}

__global__ void final_mean_bias_relu(const float* __restrict__ out2,
                                     const float* __restrict__ b2,
                                     float* __restrict__ z) {
  int i = blockIdx.x * 256 + threadIdx.x;          // over N*C
  if (i >= N_NODES * C) return;
  int n = i / C, c = i % C;
  const float* p = out2 + (size_t)n * HC + c;
  float s = 0.f;
  #pragma unroll
  for (int hh = 0; hh < HEADS; ++hh) s += p[hh * C];
  float v = s * (1.0f / HEADS) + b2[c];
  z[i] = v > 0.f ? v : 0.f;
}

// ---------------------------------------------------------------- launch
extern "C" void kernel_launch(void* const* d_in, const int* in_sizes, int n_in,
                              void* d_out, int out_size, void* d_ws, size_t ws_size,
                              hipStream_t stream) {
  (void)in_sizes; (void)n_in; (void)out_size; (void)ws_size;
  const float* x   = (const float*)d_in[0];
  const int*   ei  = (const int*)d_in[1];
  const float* W1  = (const float*)d_in[2];
  const float* as1 = (const float*)d_in[3];
  const float* ad1 = (const float*)d_in[4];
  const float* b1  = (const float*)d_in[5];
  const float* W2  = (const float*)d_in[6];
  const float* as2 = (const float*)d_in[7];
  const float* ad2 = (const float*)d_in[8];
  const float* b2  = (const float*)d_in[9];
  float* z = (float*)d_out;

  char* ws = (char*)d_ws;
  size_t off = 0;
  auto carve = [&](size_t bytes) -> void* {
    void* p = ws + off;
    off = (off + bytes + 255) & ~(size_t)255;
    return p;
  };
  unsigned short* xb    = (unsigned short*)carve((size_t)N_NODES * KP1 * 2); // also h1relu bf16
  unsigned short* wt    = (unsigned short*)carve((size_t)HC * KP1 * 2);      // W1T then W2T
  float*          hbuf  = (float*)carve((size_t)N_NODES * HC * 4);           // h1 then h2
  float*          obuf  = (float*)carve((size_t)N_NODES * HC * 4);           // out1 then out2
  float*          a_s   = (float*)carve((size_t)N_NODES * HEADS * 4);
  float*          a_d   = (float*)carve((size_t)N_NODES * HEADS * 4);
  int*            cnts  = (int*)carve((size_t)N_NODES * 4);
  int*            rowp  = (int*)carve((size_t)(N_NODES + 1) * 4);
  int*            fpos  = (int*)carve((size_t)N_NODES * 4);
  int*            csrc  = (int*)carve((size_t)ETOT * 4);

  const int mtiles = (N_NODES + 127) / 128;
  const dim3 gemm_grid(mtiles, HC / 128);
  const int wave_blocks = (N_NODES * HEADS * 32 + 255) / 256;

  // CSR for both layers (graph is identical)
  zero_i32<<<(N_NODES + 255) / 256, 256, 0, stream>>>(cnts, N_NODES);
  csr_count<<<(ETOT + 255) / 256, 256, 0, stream>>>(ei, cnts);
  scan_fill<<<1, 1024, 0, stream>>>(cnts, rowp, fpos, N_NODES);
  csr_fill<<<(ETOT + 255) / 256, 256, 0, stream>>>(ei, fpos, csrc);

  // ---- layer 1
  convert_pad_bf16<<<((size_t)N_NODES * KP1 + 255) / 256, 256, 0, stream>>>(x, xb, N_NODES, F_IN, KP1);
  transpose_w_bf16<<<((size_t)HC * KP1 + 255) / 256, 256, 0, stream>>>(W1, wt, F_IN, HC, KP1);
  wmma_gemm<<<gemm_grid, 256, 0, stream>>>(xb, wt, hbuf, N_NODES, KP1, KP1 / 32);
  att_dots<<<wave_blocks, 256, 0, stream>>>(hbuf, as1, ad1, a_s, a_d);
  gat_aggregate<<<wave_blocks, 256, 0, stream>>>(hbuf, a_s, a_d, rowp, csrc, obuf);
  bias_relu_bf16<<<((size_t)N_NODES * HC + 255) / 256, 256, 0, stream>>>(obuf, b1, xb, N_NODES * HC);

  // ---- layer 2
  transpose_w_bf16<<<((size_t)HC * K2 + 255) / 256, 256, 0, stream>>>(W2, wt, K2, HC, K2);
  wmma_gemm<<<gemm_grid, 256, 0, stream>>>(xb, wt, hbuf, N_NODES, K2, K2 / 32);
  att_dots<<<wave_blocks, 256, 0, stream>>>(hbuf, as2, ad2, a_s, a_d);
  gat_aggregate<<<wave_blocks, 256, 0, stream>>>(hbuf, a_s, a_d, rowp, csrc, obuf);

  final_mean_bias_relu<<<((size_t)N_NODES * C + 255) / 256, 256, 0, stream>>>(obuf, b2, z);
}
